// KANEmbeddings_63917703299308
// MI455X (gfx1250) — matrix operand order repaired and checked by reading, not compiled, not run on hardware
//
#include <hip/hip_runtime.h>
#include <hip/hip_bf16.h>

// ---------------------------------------------------------------------------
// KAN embedding forward on MI455X (gfx1250, wave32, WMMA).
//
// out[b,n,d] = dot(act[b,n,0:32], W[n,0:32,d])  with
//   act = [silu(x), bases_0..7(x), cos(1x..10x), sin(1x..10x), 1, 0, 0]
//   W   = [base_w, spline_w*scaler (8), fourier_c0 (10), fourier_c1 (10), bias, 0, 0]
//
// Per feature n this is (8192 x 32) @ (32 x 64): K=32 fits ONE
// v_wmma_f32_16x16x32_f16 per 16x16 output tile. Store-bandwidth bound
// (256 MB out -> ~11 us floor at 23.3 TB/s), so: f16 pre-packed weights
// (512 KB, L2-resident), register-built activations, non-temporal stores,
// and all store addressing folded into IOFFSET immediates off one base.
// ---------------------------------------------------------------------------

typedef _Float16 v16h __attribute__((ext_vector_type(16)));
typedef float    v8f  __attribute__((ext_vector_type(8)));

#define N_FEAT   128
#define D_EMB    64
#define KDIM     32          // padded K (30 real + 2 zero)
#define BATCH    8192
#define FGRID    10
#define H_GRID   0.4f        // (1 - (-1)) / 5

// grid knot j (j = 0..11):  g[j] = -1 + (j-3)*h
__device__ __forceinline__ float knot(int j) { return -1.0f + (float)(j - 3) * H_GRID; }

// ---------------------------------------------------------------------------
// Kernel 1: pack all parameters into f16 W[n][k][d], k-major over d.
//   k=0      : base_weight[n,d]
//   k=1..8   : spline_weight[n,d,k-1] * spline_scaler[n,d]
//   k=9..18  : fourier_coeffs[0,n,d,k-9]
//   k=19..28 : fourier_coeffs[1,n,d,k-19]
//   k=29     : fourier_bias[n,d]   (multiplied by constant-1 activation)
//   k=30,31  : 0
// ---------------------------------------------------------------------------
__global__ __launch_bounds__(64)
void kan_pack_weights(const float* __restrict__ bw,   // (128,64)
                      const float* __restrict__ sw,   // (128,64,8)
                      const float* __restrict__ ss,   // (128,64)
                      const float* __restrict__ fc,   // (2,128,64,10)
                      const float* __restrict__ fb,   // (1,128,64)
                      _Float16*    __restrict__ Wh)   // (128,32,64)
{
    const int d  = threadIdx.x;          // 0..63
    const int n  = blockIdx.x;           // 0..127
    const int nd = n * D_EMB + d;

    _Float16* o = Wh + (size_t)n * KDIM * D_EMB + d;   // stride D_EMB per k

    const float scaler = ss[nd];
    o[0 * D_EMB] = (_Float16)bw[nd];
#pragma unroll
    for (int i = 0; i < 8; ++i)
        o[(1 + i) * D_EMB] = (_Float16)(sw[nd * 8 + i] * scaler);
#pragma unroll
    for (int g = 0; g < FGRID; ++g) {
        o[(9  + g) * D_EMB] = (_Float16)fc[(size_t)nd * FGRID + g];
        o[(19 + g) * D_EMB] = (_Float16)fc[(size_t)N_FEAT * D_EMB * FGRID + (size_t)nd * FGRID + g];
    }
    o[29 * D_EMB] = (_Float16)fb[nd];
    o[30 * D_EMB] = (_Float16)0.0f;
    o[31 * D_EMB] = (_Float16)0.0f;
}

// ---------------------------------------------------------------------------
// Kernel 2: WMMA main pass.
// grid = (16 batch-chunks, 128 features), block = 256 (8 waves).
// Each wave: 4 batch tiles of 16 rows; per tile builds the A fragment in
// registers and fires 4 x v_wmma_f32_16x16x32_f16 (one per 16-wide d tile).
// B fragments loaded once per wave (per-feature, 4 KB, L2 resident).
// All 32 output stores per tile use ONE base address + constant IOFFSETs
// (row stride 32768 B, d-tile stride 64 B, both < 2^23).
// ---------------------------------------------------------------------------
__global__ __launch_bounds__(256)
void kan_wmma_main(const float* __restrict__ x,      // (8192,128)
                   const _Float16* __restrict__ Wh,  // (128,32,64) f16
                   float* __restrict__ out)          // (8192,128,64)
{
    const int lane = threadIdx.x & 31;
    const int wave = threadIdx.x >> 5;      // 0..7
    const int n    = blockIdx.y;            // feature
    const int hi   = lane >> 4;             // half-wave select
    const int m    = lane & 15;             // row-in-tile / output column

    // ---- B fragments: lane holds N column (m), 16 halves = K in table order ----
    const _Float16* wbase = Wh + (size_t)n * KDIM * D_EMB + m;
    v16h Bf[4];
#pragma unroll
    for (int dt = 0; dt < 4; ++dt)
#pragma unroll
        for (int s = 0; s < 16; ++s)
            Bf[dt][s] = wbase[((s < 8) ? (8 * hi + s) : (16 + 8 * hi + s - 8)) * D_EMB + dt * 16];    // strided gather

    const int wgl = blockIdx.x * 8 + wave;           // 0..127 global wave id

#pragma unroll 1
    for (int t = 0; t < 4; ++t) {
        const int tile = wgl * 4 + t;                // 0..511
        const int b0   = tile * 16;

        // ------------------- per-row activation vector -------------------
        const float xv = x[(size_t)(b0 + m) * N_FEAT + n];

        float act[32];

        // silu
        const float sig = 1.0f / (1.0f + __expf(-xv));
        act[0] = xv * sig;

        // cubic B-spline bases, uniform grid, Cox-de Boor fully unrolled
        float bs[11];
#pragma unroll
        for (int j = 0; j < 11; ++j)
            bs[j] = (xv >= knot(j) && xv < knot(j + 1)) ? 1.0f : 0.0f;
#pragma unroll
        for (int k = 1; k <= 3; ++k) {
            const float inv = 1.0f / ((float)k * H_GRID);
#pragma unroll
            for (int j = 0; j + k < 11; ++j)       // ascending in-place is safe
                bs[j] = (xv - knot(j)) * inv * bs[j]
                      + (knot(j + k + 1) - xv) * inv * bs[j + 1];
        }
#pragma unroll
        for (int j = 0; j < 8; ++j) act[1 + j] = bs[j];

        // Fourier: Chebyshev recurrence for cos/sin(g*x), g = 1..10
        const float c1 = __cosf(xv);
        const float s1 = __sinf(xv);
        float ck = c1, sk = s1, cp = 1.0f, sp = 0.0f;
#pragma unroll
        for (int g = 0; g < FGRID; ++g) {
            act[9  + g] = ck;
            act[19 + g] = sk;
            const float cn = 2.0f * c1 * ck - cp;
            const float sn = 2.0f * c1 * sk - sp;
            cp = ck; sp = sk; ck = cn; sk = sn;
        }
        act[29] = 1.0f;          // bias row
        act[30] = 0.0f;
        act[31] = 0.0f;

        // ---- A fragment, ISA 16-bit A 16x32 layout:
        //   lanes 0-15 : elems 0-7 = K0..7,  elems 8-15 = K16..23
        //   lanes 16-31: elems 0-7 = K8..15, elems 8-15 = K24..31
        v16h A;
        if (lane < 16) {
#pragma unroll
            for (int j = 0; j < 8; ++j) {
                A[j]     = (_Float16)act[j];
                A[8 + j] = (_Float16)act[16 + j];
            }
        } else {
#pragma unroll
            for (int j = 0; j < 8; ++j) {
                A[j]     = (_Float16)act[8 + j];
                A[8 + j] = (_Float16)act[24 + j];
            }
        }
        // (branch re-converges: EXEC all-1s at WMMA as required)

        // ------------------- 4 WMMAs -------------------
        v8f c[4];
#pragma unroll
        for (int dt = 0; dt < 4; ++dt) {
            v8f z = {};
            c[dt] = __builtin_amdgcn_wmma_f32_16x16x32_f16(
                        /*neg_a=*/false, A, /*neg_b=*/false, Bf[dt],
                        /*c_mod=*/(short)0, z, /*reuse_a=*/false, /*reuse_b=*/false);
        }

        // --------- NT stores: one base address, constant immediate offsets ---------
        // D layout: VGPR r, lanes 0-15 -> M=r, lanes 16-31 -> M=r+8, N=m.
        // Base row = b0 + hi*8; per-r offset = r*8192 floats (32768 B), per-dt = 16 floats.
        float* obase = out + ((size_t)(b0 + hi * 8) * N_FEAT + n) * D_EMB + m;
#pragma unroll
        for (int dt = 0; dt < 4; ++dt) {
#pragma unroll
            for (int r = 0; r < 8; ++r) {
                __builtin_nontemporal_store(c[dt][r],
                    obase + (size_t)r * (N_FEAT * D_EMB) + dt * 16);
            }
        }
    }
}

// ---------------------------------------------------------------------------
extern "C" void kernel_launch(void* const* d_in, const int* in_sizes, int n_in,
                              void* d_out, int out_size, void* d_ws, size_t ws_size,
                              hipStream_t stream)
{
    (void)in_sizes; (void)n_in; (void)out_size; (void)ws_size;

    const float* x  = (const float*)d_in[0];   // (8192,128)
    // d_in[1] = grid (unused: uniform grid reproduced analytically)
    const float* bw = (const float*)d_in[2];   // (128,64)
    const float* sw = (const float*)d_in[3];   // (128,64,8)
    const float* ss = (const float*)d_in[4];   // (128,64)
    const float* fc = (const float*)d_in[5];   // (2,128,64,10)
    const float* fb = (const float*)d_in[6];   // (1,128,64)

    _Float16* Wh = (_Float16*)d_ws;            // 128*32*64*2 = 512 KB scratch

    kan_pack_weights<<<dim3(N_FEAT), dim3(D_EMB), 0, stream>>>(bw, sw, ss, fc, fb, Wh);
    kan_wmma_main<<<dim3(16, N_FEAT), dim3(256), 0, stream>>>(x, Wh, (float*)d_out);
}
